// inner_correlation_12017318494794
// MI455X (gfx1250) — matrix operand order repaired and checked by reading, not compiled, not run on hardware
//
#include <hip/hip_runtime.h>

// Problem constants (match the reference)
#define NSEQ 128   // N
#define LIDX 32    // L (index_len)
#define NH   8     // heads
#define NE   32    // E
#define NO   32    // O
#define LPAD 33    // LDS row padding to avoid bank conflicts

typedef __attribute__((ext_vector_type(16))) _Float16 v16h;
typedef __attribute__((ext_vector_type(8)))  float    v8f;

__global__ __launch_bounds__(256) void inner_corr_kernel(
    const float* __restrict__ q, const float* __restrict__ k,
    const float* __restrict__ v, const float* __restrict__ w,
    float* __restrict__ out, int B)
{
    // sm: per-head 32x32 scratch (|S| -> softmax(|S|) -> X); Asum: head-summed scores
    __shared__ float sm[NH][LIDX][LPAD];   // ~33 KB
    __shared__ float Asum[LIDX][LPAD];     // ~4 KB

    const int bn   = blockIdx.x;           // b*NSEQ + n
    const int n    = bn % NSEQ;
    const int tid  = threadIdx.x;
    const int h    = tid >> 5;             // one wave per head
    const int lane = tid & 31;
    const int l16  = lane & 15;            // row/col within 16-wide tile
    const int kh   = lane >> 4;            // K half select (A/B operand layout)

    const size_t qkvBase = (size_t)bn * LIDX * NH * NE;  // [b][n][0][0][0]
    const float* qp = q + qkvBase;
    const float* kp = k + qkvBase;
    const float* vp = v + qkvBase;

    v16h aop[2], bop[2];
    v8f  acc[2][2];

    // ---------------- Phase 0: S_h = Q_h * K_h^T  (M=x, K=e, N=y) ----------------
    // A operand (16x32 f16): lane l%16 = row; halves i<8 -> K = kh*8+i, i>=8 -> K = 16+kh*8+(i-8)
    #pragma unroll
    for (int mt = 0; mt < 2; ++mt) {
        const int row = mt * 16 + l16;
        const float* rq = qp + ((size_t)row * NH + h) * NE;
        #pragma unroll
        for (int i = 0; i < 8; ++i) {
            aop[mt][i]     = (_Float16)rq[kh * 8 + i];
            aop[mt][i + 8] = (_Float16)rq[16 + kh * 8 + i];
        }
    }
    // B operand (32x16 f16, KxN): lane l%16 = col y; i<8 -> K = kh*8+i, i>=8 -> K = 16+kh*8+(i-8)
    #pragma unroll
    for (int nt = 0; nt < 2; ++nt) {
        const int col = nt * 16 + l16;
        const float* rk = kp + ((size_t)col * NH + h) * NE + kh * 8;
        #pragma unroll
        for (int i = 0; i < 16; ++i) bop[nt][i] = (_Float16)rk[i < 8 ? i : i + 8];
    }
    #pragma unroll
    for (int mt = 0; mt < 2; ++mt)
        #pragma unroll
        for (int nt = 0; nt < 2; ++nt) {
            v8f c = {};
            c = __builtin_amdgcn_wmma_f32_16x16x32_f16(
                    false, aop[mt], false, bop[nt], (short)0, c, false, false);
            acc[mt][nt] = c;
        }
    // Scatter |S| to LDS. C/D layout: VGPR r -> row (kh*8 + r), col = l16 within tile.
    #pragma unroll
    for (int mt = 0; mt < 2; ++mt)
        #pragma unroll
        for (int nt = 0; nt < 2; ++nt)
            #pragma unroll
            for (int r = 0; r < 8; ++r)
                sm[h][mt * 16 + kh * 8 + r][nt * 16 + l16] = fabsf(acc[mt][nt][r]);
    __syncthreads();

    // ---------------- Phase 1: row softmax, P_h = softmax(|S_h|) ----------------
    {
        float* rowp = sm[h][lane];         // row x = lane of head h
        float m = -3.0e38f;
        #pragma unroll
        for (int y = 0; y < LIDX; ++y) m = fmaxf(m, rowp[y]);
        float s = 0.f;
        #pragma unroll
        for (int y = 0; y < LIDX; ++y) { float t = __expf(rowp[y] - m); rowp[y] = t; s += t; }
        const float inv = __frcp_rn(s);
        #pragma unroll
        for (int y = 0; y < LIDX; ++y) rowp[y] *= inv;
    }
    __syncthreads();

    // ---------------- Phase 2: A[x,y] = sum_h P_h[x,y] ----------------
    for (int idx = tid; idx < LIDX * LIDX; idx += 256) {
        const int x = idx >> 5, y = idx & 31;
        float s = 0.f;
        #pragma unroll
        for (int hh = 0; hh < NH; ++hh) s += sm[hh][x][y];
        Asum[x][y] = s;
    }
    __syncthreads();

    // ---------------- Phase 3: X_h = A * V_h  (M=x, K=y, N=e) ----------------
    #pragma unroll
    for (int mt = 0; mt < 2; ++mt) {
        const int row = mt * 16 + l16;
        #pragma unroll
        for (int i = 0; i < 8; ++i) {
            aop[mt][i]     = (_Float16)Asum[row][kh * 8 + i];
            aop[mt][i + 8] = (_Float16)Asum[row][16 + kh * 8 + i];
        }
    }
    #pragma unroll
    for (int nt = 0; nt < 2; ++nt) {
        const int col = nt * 16 + l16;   // e
        #pragma unroll
        for (int i = 0; i < 16; ++i) {
            const int y = (i < 8) ? kh * 8 + i : 16 + kh * 8 + (i - 8);
            bop[nt][i] = (_Float16)vp[((size_t)y * NH + h) * NE + col];
        }
    }
    #pragma unroll
    for (int mt = 0; mt < 2; ++mt)
        #pragma unroll
        for (int nt = 0; nt < 2; ++nt) {
            v8f c = {};
            c = __builtin_amdgcn_wmma_f32_16x16x32_f16(
                    false, aop[mt], false, bop[nt], (short)0, c, false, false);
            acc[mt][nt] = c;
        }
    // X[x][e] into sm[h] (phase-2 barrier already retired all readers of sm)
    #pragma unroll
    for (int mt = 0; mt < 2; ++mt)
        #pragma unroll
        for (int nt = 0; nt < 2; ++nt)
            #pragma unroll
            for (int r = 0; r < 8; ++r)
                sm[h][mt * 16 + kh * 8 + r][nt * 16 + l16] = acc[mt][nt][r];
    __syncthreads();

    // ---------------- Phase 4: out[o,x] = sum_e X[x,e] * W[n,h,e,o,x] ----------------
    // lane = x; weights reads are 32-lane coalesced (x is fastest axis).
    const float* wh = w + (((size_t)n * NH + h) * NE) * (NO * LIDX);  // w[n][h][e][o][x]
    float*       op = out + (((size_t)bn * NH + h) * NO) * LIDX;      // out[b][n][h][o][x]
    const float* xr = sm[h][lane];                                    // X[x][*]
    for (int o = 0; o < NO; ++o) {
        float a = 0.f;
        #pragma unroll
        for (int e = 0; e < NE; ++e)
            a = fmaf(xr[e], wh[((size_t)e * NO + o) * LIDX + lane], a);
        op[(size_t)o * LIDX + lane] = a;
    }
}

extern "C" void kernel_launch(void* const* d_in, const int* in_sizes, int n_in,
                              void* d_out, int out_size, void* d_ws, size_t ws_size,
                              hipStream_t stream) {
    const float* q = (const float*)d_in[0];
    const float* k = (const float*)d_in[1];
    const float* v = (const float*)d_in[2];
    const float* w = (const float*)d_in[3];
    // d_in[4] = d_model scalar (unused; constants are compile-time)
    float* out = (float*)d_out;

    const int B = in_sizes[0] / (NSEQ * LIDX * NH * NE);   // = 16
    const int blocks = B * NSEQ;                           // one block per (b,n)

    inner_corr_kernel<<<blocks, 256, 0, stream>>>(q, k, v, w, out, B);
    (void)n_in; (void)out_size; (void)d_ws; (void)ws_size;
}
